// MhaSelfAttenLayer_7627861918007
// MI455X (gfx1250) — compile-verified, matrix operand order checked by reading, not run on hardware
//
#include <hip/hip_runtime.h>
#include <hip/hip_bf16.h>
#include <math.h>

typedef __attribute__((ext_vector_type(16))) _Float16 v16h;
typedef __attribute__((ext_vector_type(8)))  _Float16 v8h;
typedef __attribute__((ext_vector_type(8)))  float    v8f;

#define NB    8
#define TT    1024
#define CC    1024
#define NHEADS 16
#define HD    64
#define HIDD  4096

// ---------------------------------------------------------------------------
// WMMA fragment helpers (wave32, 16-bit A/B layouts per CDNA5 ISA 7.12.2)
// lanes 0-15: row=lane, K koff=0 (K 0..7 & 16..23); lanes 16-31: row=lane&15,
// koff=8 (K 8..15 & 24..31).
// ---------------------------------------------------------------------------
__device__ __forceinline__ v16h ld_frag_row(const _Float16* __restrict__ base, int ld) {
  const int lane = threadIdx.x & 31;
  const int row  = lane & 15;
  const int koff = (lane & 16) ? 8 : 0;
  const _Float16* p = base + (size_t)row * ld + koff;
  v8h lo = *(const v8h*)(p);        // K = koff .. koff+7
  v8h hi = *(const v8h*)(p + 16);   // K = 16+koff .. 23+koff
  v16h r;
#pragma unroll
  for (int i = 0; i < 8; ++i) { r[i] = lo[i]; r[i + 8] = hi[i]; }
  return r;
}

__device__ __forceinline__ v16h ld_frag_lds(const _Float16* p) {
  v8h lo = *(const v8h*)(p);
  v8h hi = *(const v8h*)(p + 16);
  v16h r;
#pragma unroll
  for (int i = 0; i < 8; ++i) { r[i] = lo[i]; r[i + 8] = hi[i]; }
  return r;
}

// B fragment where logical B[k][n] lives column-major (base[k*ld + n]).
__device__ __forceinline__ v16h ld_frag_col(const _Float16* __restrict__ base, int ld) {
  const int lane = threadIdx.x & 31;
  const int col  = lane & 15;
  const int koff = (lane & 16) ? 8 : 0;
  v16h r;
#pragma unroll
  for (int i = 0; i < 8; ++i) {
    r[i]     = base[(size_t)(koff + i) * ld + col];
    r[i + 8] = base[(size_t)(16 + koff + i) * ld + col];
  }
  return r;
}

// ---------------------------------------------------------------------------
// LDS-staged WMMA GEMM: D[M,N] = A[M,K] * B[N,K]^T (row-major f16, K contig).
// Block = 256 thr = 8 waves; block tile 128x128; wave tile 32x64.
// A/B panels double-buffered in LDS, filled with global_load_async_to_lds_b128
// (ASYNCcnt) and consumed via ds_load_b128 fragment reads.
// EPI: 0 = f16 ; 1 = f32 ; 2 = relu(x+bias)->f16 ; 3 = x+bias->f32
// ---------------------------------------------------------------------------
template <int EPI>
__global__ __launch_bounds__(256)
void gemm_wmma_kernel(const _Float16* __restrict__ A, const _Float16* __restrict__ B,
                      const float* __restrict__ bias, void* __restrict__ D,
                      int M, int N, int K) {
  __shared__ __align__(16) _Float16 As[2][128][40];   // 40-half padded rows
  __shared__ __align__(16) _Float16 Bs[2][128][40];

  const int tid  = threadIdx.x;
  const int w    = tid >> 5;
  const int wm   = w >> 1;          // 0..3 -> 32-row strip
  const int wn   = w & 1;           // 0..1 -> 64-col strip
  const int rowB = blockIdx.y * 128;
  const int colB = blockIdx.x * 128;

  v8f acc[2][4];
#pragma unroll
  for (int mi = 0; mi < 2; ++mi)
#pragma unroll
    for (int j = 0; j < 4; ++j) acc[mi][j] = (v8f){};

  // async prefetch of one 128x32 A panel + 128x32 B panel (4 b128 / thread)
  auto prefetch = [&](int buf, int k) {
#pragma unroll
    for (int t = 0; t < 2; ++t) {
      const int i   = tid + t * 256;   // 0..511
      const int row = i >> 2;
      const int seg = (i & 3) * 8;     // halves (16B chunks)
      const unsigned long long ga =
          (unsigned long long)(A + (size_t)(rowB + row) * K + k + seg);
      const unsigned la = (unsigned)(uintptr_t)&As[buf][row][seg];
      asm volatile("global_load_async_to_lds_b128 %0, %1, off"
                   :: "v"(la), "v"(ga) : "memory");
      const unsigned long long gb =
          (unsigned long long)(B + (size_t)(colB + row) * K + k + seg);
      const unsigned lb = (unsigned)(uintptr_t)&Bs[buf][row][seg];
      asm volatile("global_load_async_to_lds_b128 %0, %1, off"
                   :: "v"(lb), "v"(gb) : "memory");
    }
  };

  prefetch(0, 0);
  const int KT = K >> 5;
  for (int kt = 0; kt < KT; ++kt) {
    const int buf = kt & 1;
    if (kt + 1 < KT) {
      prefetch(buf ^ 1, (kt + 1) << 5);
      // async loads retire in order: <=4 outstanding == current panel landed
      asm volatile("s_wait_asynccnt 4" ::: "memory");
    } else {
      asm volatile("s_wait_asynccnt 0" ::: "memory");
    }
    __syncthreads();

    const int lane = tid & 31;
    const int fr   = lane & 15;
    const int koff = (lane & 16) ? 8 : 0;
    v16h a[2];
#pragma unroll
    for (int mi = 0; mi < 2; ++mi)
      a[mi] = ld_frag_lds(&As[buf][wm * 32 + mi * 16 + fr][koff]);
#pragma unroll
    for (int j = 0; j < 4; ++j) {
      v16h b = ld_frag_lds(&Bs[buf][wn * 64 + j * 16 + fr][koff]);
#pragma unroll
      for (int mi = 0; mi < 2; ++mi)
        acc[mi][j] = __builtin_amdgcn_wmma_f32_16x16x32_f16(
            false, a[mi], false, b, (short)0, acc[mi][j], false, false);
    }
    __syncthreads();   // panel fully consumed before it is overwritten
  }

  const int lane = tid & 31;
  const int colL = lane & 15;
  const int rb   = (lane & 16) ? 8 : 0;
#pragma unroll
  for (int mi = 0; mi < 2; ++mi) {
#pragma unroll
    for (int j = 0; j < 4; ++j) {
      const int col  = colB + wn * 64 + j * 16 + colL;
      const int row0 = rowB + wm * 32 + mi * 16 + rb;
#pragma unroll
      for (int r = 0; r < 8; ++r) {
        const size_t idx = (size_t)(row0 + r) * N + col;
        float v = acc[mi][j][r];
        if (EPI == 0) {
          ((_Float16*)D)[idx] = (_Float16)v;
        } else if (EPI == 1) {
          ((float*)D)[idx] = v;
        } else if (EPI == 2) {
          v += bias[col];
          ((_Float16*)D)[idx] = (_Float16)(v > 0.f ? v : 0.f);
        } else {
          ((float*)D)[idx] = v + bias[col];
        }
      }
    }
  }
}

// ---------------------------------------------------------------------------
// Flash attention: 1 wave per (batch, head, 16-query tile). Causal.
// qkv: [N*T, 3C] f16 (Q | K | V slabs). ctx out: [N*T, C] f16.
// ---------------------------------------------------------------------------
__global__ __launch_bounds__(32)
void attn_kernel(const _Float16* __restrict__ qkv, _Float16* __restrict__ ctx) {
  const int b  = blockIdx.z;
  const int h  = blockIdx.y;
  const int q0 = blockIdx.x * 16;
  const int lane = threadIdx.x & 31;
  const int colL = lane & 15;
  const int rb   = (lane & 16) ? 8 : 0;
  const int LD   = 3 * CC;
  const float scale = 0.125f;   // 1/sqrt(64)

  __shared__ _Float16 Plds[16][34];

  const size_t rowbase = (size_t)(b * TT) * LD;
  const _Float16* Qb = qkv + rowbase + (size_t)q0 * LD + h * HD;
  const v16h qf0 = ld_frag_row(Qb, LD);        // d 0..31
  const v16h qf1 = ld_frag_row(Qb + 32, LD);   // d 32..63

  v8f acc[4];
#pragma unroll
  for (int j = 0; j < 4; ++j) acc[j] = (v8f){};
  float m_run[8], l_run[8];
#pragma unroll
  for (int r = 0; r < 8; ++r) { m_run[r] = -3.0e38f; l_run[r] = 0.f; }

  for (int kb = 0; kb < q0 + 16; kb += 32) {
    const _Float16* Kb = qkv + rowbase + (size_t)kb * LD + CC + h * HD;
    v16h k00 = ld_frag_row(Kb, LD);
    v16h k01 = ld_frag_row(Kb + 32, LD);
    v16h k10 = ld_frag_row(Kb + (size_t)16 * LD, LD);
    v16h k11 = ld_frag_row(Kb + (size_t)16 * LD + 32, LD);

    v8f s0 = (v8f){}, s1 = (v8f){};
    s0 = __builtin_amdgcn_wmma_f32_16x16x32_f16(false, qf0, false, k00, (short)0, s0, false, false);
    s0 = __builtin_amdgcn_wmma_f32_16x16x32_f16(false, qf1, false, k01, (short)0, s0, false, false);
    s1 = __builtin_amdgcn_wmma_f32_16x16x32_f16(false, qf0, false, k10, (short)0, s1, false, false);
    s1 = __builtin_amdgcn_wmma_f32_16x16x32_f16(false, qf1, false, k11, (short)0, s1, false, false);

    float p0[8], p1[8], mb[8];
#pragma unroll
    for (int r = 0; r < 8; ++r) {
      const int q = q0 + rb + r;
      float a = s0[r] * scale; if (kb + colL      > q) a = -3.0e38f;
      float c = s1[r] * scale; if (kb + 16 + colL > q) c = -3.0e38f;
      p0[r] = a; p1[r] = c;
      mb[r] = fmaxf(a, c);
    }
#pragma unroll
    for (int off = 1; off <= 8; off <<= 1)
#pragma unroll
      for (int r = 0; r < 8; ++r) mb[r] = fmaxf(mb[r], __shfl_xor(mb[r], off, 32));

    float alpha[8], rs[8];
#pragma unroll
    for (int r = 0; r < 8; ++r) {
      const float mn = fmaxf(m_run[r], mb[r]);
      alpha[r] = __expf(m_run[r] - mn);
      m_run[r] = mn;
      p0[r] = __expf(p0[r] - mn);
      p1[r] = __expf(p1[r] - mn);
      rs[r] = p0[r] + p1[r];
    }
#pragma unroll
    for (int off = 1; off <= 8; off <<= 1)
#pragma unroll
      for (int r = 0; r < 8; ++r) rs[r] += __shfl_xor(rs[r], off, 32);
#pragma unroll
    for (int r = 0; r < 8; ++r) l_run[r] = l_run[r] * alpha[r] + rs[r];
#pragma unroll
    for (int j = 0; j < 4; ++j)
#pragma unroll
      for (int r = 0; r < 8; ++r) acc[j][r] *= alpha[r];

    __syncthreads();
#pragma unroll
    for (int r = 0; r < 8; ++r) {
      Plds[rb + r][colL]      = (_Float16)p0[r];
      Plds[rb + r][16 + colL] = (_Float16)p1[r];
    }
    __syncthreads();

    v16h pf;
    {
      const int row  = colL;
      const int koff = (lane & 16) ? 8 : 0;
#pragma unroll
      for (int i = 0; i < 8; ++i) {
        pf[i]     = Plds[row][koff + i];
        pf[i + 8] = Plds[row][16 + koff + i];
      }
    }

    const _Float16* Vb = qkv + rowbase + (size_t)kb * LD + 2 * CC + h * HD;
#pragma unroll
    for (int j = 0; j < 4; ++j) {
      v16h vf = ld_frag_col(Vb + j * 16, LD);
      acc[j] = __builtin_amdgcn_wmma_f32_16x16x32_f16(false, pf, false, vf,
                                                      (short)0, acc[j], false, false);
    }
  }

  float inv[8];
#pragma unroll
  for (int r = 0; r < 8; ++r) inv[r] = 1.f / l_run[r];
#pragma unroll
  for (int j = 0; j < 4; ++j)
#pragma unroll
    for (int r = 0; r < 8; ++r)
      ctx[(size_t)(b * TT + q0 + rb + r) * CC + h * HD + j * 16 + colL] =
          (_Float16)(acc[j][r] * inv[r]);
}

// ---------------------------------------------------------------------------
// Elementwise / reduction kernels
// ---------------------------------------------------------------------------
__global__ __launch_bounds__(256)
void prep_kernel(const float* __restrict__ x, float* __restrict__ src,
                 _Float16* __restrict__ xq) {
  const int row = blockIdx.x;   // n*T + t
  const int n = row >> 10;
  const int t = row & 1023;
  const float k0 = -logf(10000.0f) / (float)CC;
  for (int c = threadIdx.x; c < CC; c += 256) {
    const float xv = x[((size_t)n * CC + c) * TT + t];
    src[(size_t)row * CC + c] = xv;
    const int i2 = c & ~1;
    const float ang = (float)t * __expf(k0 * (float)i2);
    const float pe = (c & 1) ? __cosf(ang) : __sinf(ang);
    xq[(size_t)row * CC + c] = (_Float16)(xv + pe);
  }
}

__global__ void cvt_f32_f16(const float* __restrict__ in, _Float16* __restrict__ out, int n) {
  const int i = blockIdx.x * blockDim.x + threadIdx.x;
  if (i < n) out[i] = (_Float16)in[i];
}

__device__ __forceinline__ float block_sum_256(float v, float* red) {
  const int tid = threadIdx.x;
  red[tid] = v; __syncthreads();
  for (int o = 128; o > 0; o >>= 1) {
    if (tid < o) red[tid] += red[tid + o];
    __syncthreads();
  }
  const float s = red[0];
  __syncthreads();
  return s;
}

__global__ __launch_bounds__(256)
void ln1_kernel(const float* __restrict__ a, const float* __restrict__ bb,
                const float* __restrict__ g, const float* __restrict__ beta,
                float* __restrict__ hout, _Float16* __restrict__ h16) {
  __shared__ float red[256];
  const int row = blockIdx.x, tid = threadIdx.x;
  float vals[4], s = 0.f;
#pragma unroll
  for (int i = 0; i < 4; ++i) {
    const int c = tid + i * 256;
    const float v = a[(size_t)row * CC + c] + bb[(size_t)row * CC + c];
    vals[i] = v; s += v;
  }
  const float mu = block_sum_256(s, red) * (1.f / CC);
  float sq = 0.f;
#pragma unroll
  for (int i = 0; i < 4; ++i) { const float d = vals[i] - mu; sq += d * d; }
  const float rstd = rsqrtf(block_sum_256(sq, red) * (1.f / CC) + 1e-5f);
#pragma unroll
  for (int i = 0; i < 4; ++i) {
    const int c = tid + i * 256;
    const float o = (vals[i] - mu) * rstd * g[c] + beta[c];
    hout[(size_t)row * CC + c] = o;
    h16[(size_t)row * CC + c] = (_Float16)o;
  }
}

__global__ __launch_bounds__(256)
void ln2t_kernel(const float* __restrict__ a, const float* __restrict__ bb,
                 const float* __restrict__ g, const float* __restrict__ beta,
                 float* __restrict__ out) {
  __shared__ float red[256];
  const int row = blockIdx.x, tid = threadIdx.x;
  const int n = row >> 10, t = row & 1023;
  float vals[4], s = 0.f;
#pragma unroll
  for (int i = 0; i < 4; ++i) {
    const int c = tid + i * 256;
    const float v = a[(size_t)row * CC + c] + bb[(size_t)row * CC + c];
    vals[i] = v; s += v;
  }
  const float mu = block_sum_256(s, red) * (1.f / CC);
  float sq = 0.f;
#pragma unroll
  for (int i = 0; i < 4; ++i) { const float d = vals[i] - mu; sq += d * d; }
  const float rstd = rsqrtf(block_sum_256(sq, red) * (1.f / CC) + 1e-5f);
#pragma unroll
  for (int i = 0; i < 4; ++i) {
    const int c = tid + i * 256;
    out[((size_t)n * CC + c) * TT + t] = (vals[i] - mu) * rstd * g[c] + beta[c];
  }
}

// ---------------------------------------------------------------------------
extern "C" void kernel_launch(void* const* d_in, const int* in_sizes, int n_in,
                              void* d_out, int out_size, void* d_ws, size_t ws_size,
                              hipStream_t stream) {
  const float* x    = (const float*)d_in[0];
  const float* w_in = (const float*)d_in[1];
  const float* w_o  = (const float*)d_in[2];
  const float* ln1g = (const float*)d_in[3];
  const float* ln1b = (const float*)d_in[4];
  const float* ln2g = (const float*)d_in[5];
  const float* ln2b = (const float*)d_in[6];
  const float* w1   = (const float*)d_in[7];
  const float* b1   = (const float*)d_in[8];
  const float* w2   = (const float*)d_in[9];
  const float* b2   = (const float*)d_in[10];

  char* ws = (char*)d_ws;
  const size_t MB = 1ull << 20;
  float*    src     = (float*)(ws + 0);            // 32 MiB; dead after LN1
  float*    ff2     = src;                         // alias (written by FFN2)
  float*    hbuf    = (float*)(ws + 32 * MB);      // 32 MiB
  _Float16* h16     = (_Float16*)(ws + 64 * MB);   // 16 MiB
  _Float16* xq      = (_Float16*)(ws + 80 * MB);   // 16 MiB; dead after QKV GEMM
  _Float16* ctx     = xq;                          // alias
  float*    attnout = (float*)(ws + 96 * MB);      // 32 MiB
  _Float16* qkv     = (_Float16*)(ws + 128 * MB);  // 48 MiB; dead after attention
  _Float16* ff      = qkv;                         // alias (64 MiB)
  _Float16* wqkv16  = (_Float16*)(ws + 192 * MB);
  _Float16* wout16  = wqkv16 + (size_t)3072 * 1024;
  _Float16* w116    = wout16 + (size_t)1024 * 1024;
  _Float16* w216    = w116 + (size_t)HIDD * 1024;

  // weight conversions f32 -> f16
  cvt_f32_f16<<<(3072 * 1024 + 255) / 256, 256, 0, stream>>>(w_in, wqkv16, 3072 * 1024);
  cvt_f32_f16<<<(1024 * 1024 + 255) / 256, 256, 0, stream>>>(w_o, wout16, 1024 * 1024);
  cvt_f32_f16<<<(HIDD * 1024 + 255) / 256, 256, 0, stream>>>(w1, w116, HIDD * 1024);
  cvt_f32_f16<<<(HIDD * 1024 + 255) / 256, 256, 0, stream>>>(w2, w216, HIDD * 1024);

  // transpose + PE
  prep_kernel<<<NB * TT, 256, 0, stream>>>(x, src, xq);

  // QKV projection: [8192,1024] x [3072,1024]^T -> f16 [8192,3072]
  gemm_wmma_kernel<0><<<dim3(3072 / 128, 8192 / 128), 256, 0, stream>>>(
      xq, wqkv16, nullptr, qkv, 8192, 3072, 1024);

  // causal flash attention
  attn_kernel<<<dim3(TT / 16, NHEADS, NB), 32, 0, stream>>>(qkv, ctx);

  // out projection -> f32 [8192,1024]
  gemm_wmma_kernel<1><<<dim3(1024 / 128, 8192 / 128), 256, 0, stream>>>(
      ctx, wout16, nullptr, attnout, 8192, 1024, 1024);

  // residual + LN1 -> h (f32 + f16)
  ln1_kernel<<<NB * TT, 256, 0, stream>>>(src, attnout, ln1g, ln1b, hbuf, h16);

  // FFN1: relu(h @ w1^T + b1) -> f16 [8192,4096]
  gemm_wmma_kernel<2><<<dim3(HIDD / 128, 8192 / 128), 256, 0, stream>>>(
      h16, w116, b1, ff, 8192, HIDD, 1024);

  // FFN2: ff @ w2^T + b2 -> f32 [8192,1024]
  gemm_wmma_kernel<3><<<dim3(1024 / 128, 8192 / 128), 256, 0, stream>>>(
      ff, w216, b2, ff2, 8192, 1024, HIDD);

  // residual + LN2 + transpose back to [N,C,T]
  ln2t_kernel<<<NB * TT, 256, 0, stream>>>(hbuf, ff2, ln2g, ln2b, (float*)d_out);
}
